// GraphSAGEWithAttention_9311489098207
// MI455X (gfx1250) — matrix operand order repaired and checked by reading, not compiled, not run on hardware
//
#include <hip/hip_runtime.h>
#include <hip/hip_bf16.h>
#include <math.h>

typedef __attribute__((ext_vector_type(16))) _Float16 v16h;
typedef __attribute__((ext_vector_type(8)))  float    v8f;

#define NN   20000
#define NE   320000
#define ETOT (NE + NN)   // edges + self loops
#define DIN  128
#define HID  256

// ---------------------------------------------------------------- utilities
__global__ void fill_f32(float* __restrict__ p, float v, int n) {
    int i = blockIdx.x * blockDim.x + threadIdx.x;
    if (i < n) p[i] = v;
}

__global__ void cvt_f32_to_f16(const float* __restrict__ src,
                               _Float16* __restrict__ dst, int n) {
    int i = blockIdx.x * blockDim.x + threadIdx.x;
    if (i < n) dst[i] = (_Float16)src[i];
}

// W [K, Nc] f32  ->  WT [Nc, K] f16
__global__ void transpose_cvt_f16(const float* __restrict__ W,
                                  _Float16* __restrict__ WT, int K, int Nc) {
    int i = blockIdx.x * blockDim.x + threadIdx.x;
    if (i >= K * Nc) return;
    int k = i / Nc, nc = i % Nc;
    WT[(size_t)nc * K + k] = (_Float16)W[i];
}

__device__ __forceinline__ float atomic_max_f32(float* addr, float v) {
    // signed-int max for v>=0, unsigned-int min for v<0; correct with -inf init
    if (v >= 0.0f)
        return __int_as_float(atomicMax((int*)addr, __float_as_int(v)));
    return __uint_as_float(atomicMin((unsigned int*)addr, __float_as_uint(v)));
}

// ------------------------------------------------------------- WMMA GEMM
// C[M,Nc] = A[M,K](f16) * B[K,Nc](f16, supplied as BT[Nc,K]); f32 accumulate.
// One wave32 produces a 16x64 strip (4 n-tiles). Per k-step: issue ALL 10
// b128 operand loads first (A + 4 B tiles, distinct registers), then the 4
// WMMAs, so the backend can use partial s_wait_loadcnt and overlap issue.
__global__ void __launch_bounds__(128)
gemm_wmma_f16(const _Float16* __restrict__ A,
              const _Float16* __restrict__ BT,
              float* __restrict__ C,
              int M, int K, int Nc) {
    const int lane = threadIdx.x & 31;
    const int wave = threadIdx.x >> 5;
    const int nstrips = Nc >> 6;                       // 64-wide strips
    const int totalWT  = (M >> 4) * nstrips;
    const int wt = blockIdx.x * 4 + wave;
    if (wt >= totalWT) return;
    const int mt = wt / nstrips;
    const int ntBase = (wt - mt * nstrips) * 4;        // first 16-col tile

    const int l15 = lane & 15;
    const int hb  = (lane >> 4);                       // 0 or 1 (lane half)
    const _Float16* arow = A + (size_t)(mt * 16 + l15) * K;   // A row M=lane&15
    const _Float16* brow[4];
#pragma unroll
    for (int j = 0; j < 4; ++j)                        // B col N=lane&15 of tile j
        brow[j] = BT + (size_t)((ntBase + j) * 16 + l15) * K;

    v8f acc[4] = {};
    for (int k0 = 0; k0 < K; k0 += 32) {
        union U { v16h v; uint4 u[2]; };
        U a, b[4];
        // A 16x32 layout: halfs [k0+8*hb .. +7] then [k0+16+8*hb .. +7]
        a.u[0] = *(const uint4*)(arow + k0 + 8 * hb);
        a.u[1] = *(const uint4*)(arow + k0 + 16 + 8 * hb);
#pragma unroll
        for (int j = 0; j < 4; ++j) {
            // B 32x16 layout: halfs [k0+8*hb .. +7] then [k0+16+8*hb .. +7]
            b[j].u[0] = *(const uint4*)(brow[j] + k0 + 8 * hb);
            b[j].u[1] = *(const uint4*)(brow[j] + k0 + 16 + 8 * hb);
        }
#pragma unroll
        for (int j = 0; j < 4; ++j)
            acc[j] = __builtin_amdgcn_wmma_f32_16x16x32_f16(
                         false, a.v, false, b[j].v, (short)0, acc[j], false, false);
    }
    // C/D: VGPR v -> row v + 8*hb, col lane&15
    const int rofs = hb * 8;
#pragma unroll
    for (int j = 0; j < 4; ++j) {
        float* crow = C + (size_t)(mt * 16) * Nc + (ntBase + j) * 16 + l15;
#pragma unroll
        for (int v = 0; v < 8; ++v)
            crow[(size_t)(v + rofs) * Nc] = acc[j][v];
    }
}

// ------------------------------------------------- per-node attention scores
// ls[n,h] = dot(H[n, h*hd : (h+1)*hd], a_src[h]);  same for ld
__global__ void attn_scores(const float* __restrict__ H,
                            const float* __restrict__ a_src,
                            const float* __restrict__ a_dst,
                            float* __restrict__ ls, float* __restrict__ ld,
                            int n, int heads, int hd) {
    int i = blockIdx.x * blockDim.x + threadIdx.x;
    if (i >= n * heads) return;
    int node = i / heads, h = i - node * heads;
    const float* hp = H + (size_t)node * heads * hd + h * hd;
    const float* as = a_src + h * hd;
    const float* ad = a_dst + h * hd;
    float s = 0.f, d = 0.f;
    for (int c = 0; c < hd; c += 4) {
        float4 hv = *(const float4*)(hp + c);
        float4 av = *(const float4*)(as + c);
        float4 dv = *(const float4*)(ad + c);
        s += hv.x * av.x + hv.y * av.y + hv.z * av.z + hv.w * av.w;
        d += hv.x * dv.x + hv.y * dv.y + hv.z * dv.z + hv.w * dv.w;
    }
    ls[i] = s;
    ld[i] = d;
}

// ------------------------------------------------------- edge softmax passes
__global__ void edge_logit_max(const int* __restrict__ ei, int E, int heads,
                               int total, const float* __restrict__ ls,
                               const float* __restrict__ ld,
                               float* __restrict__ Lbuf, float* __restrict__ m) {
    int i = blockIdx.x * blockDim.x + threadIdx.x;
    if (i >= total) return;
    int e = i / heads, h = i - e * heads;
    int s, d;
    if (e < E) { s = ei[e]; d = ei[E + e]; } else { s = d = e - E; }
    float v = ls[s * heads + h] + ld[d * heads + h];
    v = v > 0.f ? v : 0.2f * v;          // leaky_relu(0.2)
    Lbuf[i] = v;
    atomic_max_f32(&m[d * heads + h], v);
}

__global__ void edge_exp_denom(const int* __restrict__ ei, int E, int heads,
                               int total, float* __restrict__ Lbuf,
                               const float* __restrict__ m,
                               float* __restrict__ denom) {
    int i = blockIdx.x * blockDim.x + threadIdx.x;
    if (i >= total) return;
    int e = i / heads, h = i - e * heads;
    int d = (e < E) ? ei[E + e] : (e - E);
    float v = __expf(Lbuf[i] - m[d * heads + h]);
    Lbuf[i] = v;
    atomicAdd(&denom[d * heads + h], v);
}

// one block (HID threads) per edge: channel c handled by lane c;
// per-head alpha computed once per block and broadcast via LDS
__global__ void __launch_bounds__(HID)
edge_aggregate(const int* __restrict__ ei, int E, int heads, int headShift,
               const float* __restrict__ Lbuf, const float* __restrict__ denom,
               const float* __restrict__ H, float* __restrict__ ACC) {
    int e = blockIdx.x;
    int c = threadIdx.x;
    __shared__ int sd[2];
    __shared__ float salpha[4];
    if (c == 0) {
        if (e < E) { sd[0] = ei[e]; sd[1] = ei[E + e]; }
        else       { sd[0] = sd[1] = e - E; }
    }
    __syncthreads();
    int s = sd[0], d = sd[1];
    if (c < heads)
        salpha[c] = Lbuf[(size_t)e * heads + c] / denom[d * heads + c];
    __syncthreads();
    float alpha = salpha[c >> headShift];
    atomicAdd(&ACC[(size_t)d * HID + c], H[(size_t)s * HID + c] * alpha);
}

// bias + optional ELU; emit f32 features and f16 copy for next GEMM
__global__ void finalize_feat(const float* __restrict__ ACC,
                              const float* __restrict__ b,
                              float* __restrict__ Hout,
                              _Float16* __restrict__ Hh,
                              int total, int ncols, int doElu) {
    int i = blockIdx.x * blockDim.x + threadIdx.x;
    if (i >= total) return;
    int c = i % ncols;
    float v = ACC[i] + b[c];
    if (doElu) v = v > 0.f ? v : (__expf(v) - 1.0f);
    Hout[i] = v;
    Hh[i]   = (_Float16)v;
}

// h @ Wc[256,2] + bc, then log_softmax over the 2 classes
__global__ void classifier_lsm(const float* __restrict__ H,
                               const float* __restrict__ Wc,
                               const float* __restrict__ bc,
                               float* __restrict__ out, int n) {
    int i = blockIdx.x * blockDim.x + threadIdx.x;
    if (i >= n) return;
    const float* hp = H + (size_t)i * HID;
    float z0 = bc[0], z1 = bc[1];
    for (int c = 0; c < HID; c += 4) {
        float4 hv = *(const float4*)(hp + c);
        z0 += hv.x * Wc[(c + 0) * 2] + hv.y * Wc[(c + 1) * 2] +
              hv.z * Wc[(c + 2) * 2] + hv.w * Wc[(c + 3) * 2];
        z1 += hv.x * Wc[(c + 0) * 2 + 1] + hv.y * Wc[(c + 1) * 2 + 1] +
              hv.z * Wc[(c + 2) * 2 + 1] + hv.w * Wc[(c + 3) * 2 + 1];
    }
    float mx  = fmaxf(z0, z1);
    float lse = mx + __logf(__expf(z0 - mx) + __expf(z1 - mx));
    out[i * 2 + 0] = z0 - lse;
    out[i * 2 + 1] = z1 - lse;
}

// ---------------------------------------------------------------- launcher
extern "C" void kernel_launch(void* const* d_in, const int* in_sizes, int n_in,
                              void* d_out, int out_size, void* d_ws, size_t ws_size,
                              hipStream_t stream) {
    (void)in_sizes; (void)n_in; (void)out_size; (void)ws_size;
    const float* x   = (const float*)d_in[0];
    const int*   ei  = (const int*)  d_in[1];
    const float* W1  = (const float*)d_in[2];
    const float* a1s = (const float*)d_in[3];
    const float* a1d = (const float*)d_in[4];
    const float* b1  = (const float*)d_in[5];
    const float* W2  = (const float*)d_in[6];
    const float* a2s = (const float*)d_in[7];
    const float* a2d = (const float*)d_in[8];
    const float* b2  = (const float*)d_in[9];
    const float* W3  = (const float*)d_in[10];
    const float* a3s = (const float*)d_in[11];
    const float* a3d = (const float*)d_in[12];
    const float* b3  = (const float*)d_in[13];
    const float* Wc  = (const float*)d_in[14];
    const float* bc  = (const float*)d_in[15];
    float* out = (float*)d_out;

    // workspace layout (~58 MB), 256B-aligned slices
    char* base = (char*)d_ws;
    size_t off = 0;
    auto take = [&](size_t bytes) -> void* {
        void* p = base + off;
        off += (bytes + 255) & ~(size_t)255;
        return p;
    };
    _Float16* Hh   = (_Float16*)take((size_t)NN * HID * sizeof(_Float16));
    _Float16* WT   = (_Float16*)take((size_t)HID * HID * sizeof(_Float16));
    float*    H    = (float*)take((size_t)NN * HID * sizeof(float));
    float*    ACC  = (float*)take((size_t)NN * HID * sizeof(float));
    float*    ls   = (float*)take((size_t)NN * 4 * sizeof(float));
    float*    ld   = (float*)take((size_t)NN * 4 * sizeof(float));
    float*    mbuf = (float*)take((size_t)NN * 4 * sizeof(float));
    float*    den  = (float*)take((size_t)NN * 4 * sizeof(float));
    float*    Lbuf = (float*)take((size_t)ETOT * 4 * sizeof(float));

    // 4 waves / block, each wave a 16x64 strip: (1250 * 4) wave-tiles
    const int gemmGrid = ((NN / 16) * (HID / 64) + 3) / 4;

    auto run_layer = [&](int K, const float* W, const float* as_, const float* ad_,
                         const float* bias, int heads, int doElu) {
        const int hd = HID / heads;
        const int headShift = (heads == 4) ? 6 : 8;
        // W -> WT (f16, transposed)
        int nW = K * HID;
        hipLaunchKernelGGL(transpose_cvt_f16, dim3((nW + 255) / 256), dim3(256),
                           0, stream, W, WT, K, HID);
        // H = Hh @ W  (WMMA)
        hipLaunchKernelGGL(gemm_wmma_f16, dim3(gemmGrid), dim3(128),
                           0, stream, Hh, WT, H, NN, K, HID);
        // per-node attention scores
        int nsc = NN * heads;
        hipLaunchKernelGGL(attn_scores, dim3((nsc + 255) / 256), dim3(256),
                           0, stream, H, as_, ad_, ls, ld, NN, heads, hd);
        // init segment-max / denom / accumulator
        hipLaunchKernelGGL(fill_f32, dim3((nsc + 255) / 256), dim3(256),
                           0, stream, mbuf, -INFINITY, nsc);
        hipLaunchKernelGGL(fill_f32, dim3((nsc + 255) / 256), dim3(256),
                           0, stream, den, 0.0f, nsc);
        int nacc = NN * HID;
        hipLaunchKernelGGL(fill_f32, dim3((nacc + 255) / 256), dim3(256),
                           0, stream, ACC, 0.0f, nacc);
        // edge softmax + aggregation
        int nel = ETOT * heads;
        hipLaunchKernelGGL(edge_logit_max, dim3((nel + 255) / 256), dim3(256),
                           0, stream, ei, NE, heads, nel, ls, ld, Lbuf, mbuf);
        hipLaunchKernelGGL(edge_exp_denom, dim3((nel + 255) / 256), dim3(256),
                           0, stream, ei, NE, heads, nel, Lbuf, mbuf, den);
        hipLaunchKernelGGL(edge_aggregate, dim3(ETOT), dim3(HID),
                           0, stream, ei, NE, heads, headShift, Lbuf, den, H, ACC);
        // bias (+ELU), produce f32 features + f16 copy for next GEMM
        hipLaunchKernelGGL(finalize_feat, dim3((nacc + 255) / 256), dim3(256),
                           0, stream, ACC, bias, H, Hh, nacc, HID, doElu);
    };

    // x (f32) -> f16 GEMM operand, layout [NN, DIN]
    int nx = NN * DIN;
    hipLaunchKernelGGL(cvt_f32_to_f16, dim3((nx + 255) / 256), dim3(256),
                       0, stream, x, Hh, nx);

    run_layer(DIN, W1, a1s, a1d, b1, 4, 1);   // GAT layer 1 (concat, ELU)
    run_layer(HID, W2, a2s, a2d, b2, 4, 1);   // GAT layer 2 (concat, ELU)
    run_layer(HID, W3, a3s, a3d, b3, 1, 0);   // GAT layer 3 (1 head, mean==id)

    hipLaunchKernelGGL(classifier_lsm, dim3((NN + 255) / 256), dim3(256),
                       0, stream, H, Wc, bc, out, NN);
}
